// TriVeinAttention_75316546503106
// MI455X (gfx1250) — compile-verified, matrix operand order checked by reading, not run on hardware
//
#include <hip/hip_runtime.h>
#include <hip/hip_bf16.h>

typedef __attribute__((ext_vector_type(16))) _Float16 v16h;
typedef __attribute__((ext_vector_type(8)))  _Float16 v8h;
typedef __attribute__((ext_vector_type(8)))  float    v8f;

#define B_SZ    2
#define T_SEQ   2048
#define D_MODEL 1024
#define H_HEADS 16
#define HD      64
#define RANK    32
#define M_ROWS  (B_SZ * T_SEQ)     /* 4096 */
#define N_LR    (H_HEADS * RANK)   /* 512  */

// ---------------------------------------------------------------------------
// helpers
// ---------------------------------------------------------------------------
__device__ __forceinline__ v8f vzero8() {
  v8f z;
#pragma unroll
  for (int i = 0; i < 8; ++i) z[i] = 0.0f;
  return z;
}

// Load a 16x32 f16 A/B fragment slice for this lane: two contiguous b128 loads.
// Lane-half selects K {0..7,16..23} (half 0) vs {8..15,24..31} (half 1); caller
// passes a pointer already offset by half*8 halfs.
__device__ __forceinline__ v16h load_frag(const _Float16* p) {
  union { v16h v; v8h h[2]; } u;
  u.h[0] = *(const v8h*)p;
  u.h[1] = *(const v8h*)(p + 16);
  return u.v;
}

__device__ __forceinline__ v8f wmma16(v16h a, v16h b, v8f c) {
  return __builtin_amdgcn_wmma_f32_16x16x32_f16(false, a, false, b, (short)0, c,
                                                false, false);
}

// Load one k-chunk of GEMM fragments: A row fragment + 4 B (N-major) fragments.
__device__ __forceinline__ void load_ab(const _Float16* __restrict__ A,
                                        const _Float16* __restrict__ Wn, int arow,
                                        int nbase, int k0, int ld, int half,
                                        int l16, v16h& a, v16h* b) {
  a = load_frag(A + (size_t)arow * ld + k0 + half * 8);
#pragma unroll
  for (int t = 0; t < 4; ++t)
    b[t] = load_frag(Wn + (size_t)(nbase + t * 16 + l16) * ld + k0 + half * 8);
}

// ---------------------------------------------------------------------------
// prep kernels: f32->f16 pack and weight fusion
// ---------------------------------------------------------------------------
__global__ void cvt_f32_f16(const float* __restrict__ in, _Float16* __restrict__ out,
                            int n) {
  int i = blockIdx.x * blockDim.x + threadIdx.x;
  if (i < n) out[i] = (_Float16)in[i];
}

// Wqv_nmajor[h*32+r][d] = sum_c V[c][r] * W[h*64+c][d]   (512 x 1024 f16)
__global__ void make_wproj(const float* __restrict__ W, const float* __restrict__ V,
                           _Float16* __restrict__ out) {
  int idx = blockIdx.x * blockDim.x + threadIdx.x;  // 512*1024
  int n = idx >> 10;
  int d = idx & 1023;
  int h = n >> 5, r = n & 31;
  float s = 0.0f;
#pragma unroll 8
  for (int c = 0; c < HD; ++c)
    s += V[c * RANK + r] * W[(size_t)(h * HD + c) * D_MODEL + d];
  out[idx] = (_Float16)s;
}

// Wuo_nmajor[e][h*32+r] = sum_c U[c][r] * Wo[e][h*64+c]   (1024 x 512 f16)
__global__ void make_wout(const float* __restrict__ Wo, const float* __restrict__ U,
                          _Float16* __restrict__ out) {
  int idx = blockIdx.x * blockDim.x + threadIdx.x;  // 1024*512
  int e = idx >> 9;
  int n = idx & 511;
  int h = n >> 5, r = n & 31;
  float s = 0.0f;
#pragma unroll 8
  for (int c = 0; c < HD; ++c)
    s += U[c * RANK + r] * Wo[(size_t)e * D_MODEL + h * HD + c];
  out[idx] = (_Float16)s;
}

// projected bias zb[h*32+r] = sum_c b[h*64+c] * V[c][r]
__global__ void make_pbias(const float* __restrict__ b, const float* __restrict__ V,
                           float* __restrict__ zb) {
  int n = blockIdx.x * blockDim.x + threadIdx.x;
  if (n < N_LR) {
    int h = n >> 5, r = n & 31;
    float s = 0.0f;
#pragma unroll 8
    for (int c = 0; c < HD; ++c) s += b[h * HD + c] * V[c * RANK + r];
    zb[n] = s;
  }
}

// ---------------------------------------------------------------------------
// projection GEMM: Z[4096x512] = X16[4096x1024] @ Wn^T + zb   (Wn is N-major)
// one wave = 16x64 tile; K=1024, double-buffered 32-chunks (ping-pong).
// transposed==0 -> store zq/zk as [B,H,T,rank]; ==1 -> store zv as [B,H,rank,T]
// ---------------------------------------------------------------------------
__global__ __launch_bounds__(256) void proj_gemm(
    const _Float16* __restrict__ A, const _Float16* __restrict__ Wn,
    const float* __restrict__ zb, _Float16* __restrict__ out, int transposed) {
  const int lane = threadIdx.x & 31;
  const int wid  = threadIdx.x >> 5;
  const int gw   = blockIdx.x * 8 + wid;
  const int wn   = gw & 7;    // 8 n-tiles of 64  (N=512)
  const int wm   = gw >> 3;   // 256 m-tiles of 16
  const int half = lane >> 4, l16 = lane & 15;
  const int mbase = wm * 16, nbase = wn * 64;
  const int arow = mbase + l16;

  v8f acc[4];
#pragma unroll
  for (int t = 0; t < 4; ++t) acc[t] = vzero8();

  v16h aP, bP[4], aQ, bQ[4];
  load_ab(A, Wn, arow, nbase, 0, D_MODEL, half, l16, aP, bP);
  for (int k0 = 0; k0 < D_MODEL; k0 += 64) {
    load_ab(A, Wn, arow, nbase, k0 + 32, D_MODEL, half, l16, aQ, bQ);
#pragma unroll
    for (int t = 0; t < 4; ++t) acc[t] = wmma16(aP, bP[t], acc[t]);
    int k2 = k0 + 64;
    if (k2 >= D_MODEL) k2 = 0;  // wrapped redundant prefetch on final iter
    load_ab(A, Wn, arow, nbase, k2, D_MODEL, half, l16, aP, bP);
#pragma unroll
    for (int t = 0; t < 4; ++t) acc[t] = wmma16(aQ, bQ[t], acc[t]);
  }

#pragma unroll
  for (int t = 0; t < 4; ++t) {
    int n = nbase + t * 16 + l16;
    float bias = zb[n];
    int h = n >> 5, r = n & 31;
#pragma unroll
    for (int i = 0; i < 8; ++i) {
      int mi = mbase + i + 8 * half;
      int b = mi >> 11;         // / 2048
      int tt = mi & 2047;
      size_t addr;
      if (!transposed)
        addr = (((size_t)b * H_HEADS + h) * T_SEQ + tt) * RANK + r;
      else
        addr = (((size_t)b * H_HEADS + h) * RANK + r) * T_SEQ + tt;
      out[addr] = (_Float16)(acc[t][i] + bias);
    }
  }
}

// ---------------------------------------------------------------------------
// flash attention in rank-32 space. One wave = 16 queries of one (b,h).
// 32-key chunks: 2 score WMMAs (K=rank=32) + 2 P@zv WMMAs; kv fragments are
// software-pipelined so softmax VALU + LDS transpose overlap the next loads.
// ---------------------------------------------------------------------------
__device__ __forceinline__ void load_kv(const _Float16* __restrict__ zk_bh,
                                        const _Float16* __restrict__ zv_bh, int kb,
                                        int half, int l16, v16h& k0, v16h& k1,
                                        v16h& v0, v16h& v1) {
  k0 = load_frag(zk_bh + (size_t)(kb + l16) * RANK + half * 8);
  k1 = load_frag(zk_bh + (size_t)(kb + 16 + l16) * RANK + half * 8);
  v0 = load_frag(zv_bh + (size_t)l16 * T_SEQ + kb + half * 8);
  v1 = load_frag(zv_bh + (size_t)(16 + l16) * T_SEQ + kb + half * 8);
}

__device__ __forceinline__ void attn_step(const v16h& a_q, const v16h& bk0,
                                          const v16h& bk1, const v16h& bv0,
                                          const v16h& bv1, v8f& acc0, v8f& acc1,
                                          float* rmax, float* rsum,
                                          _Float16 (*stg)[40], int half, int l16) {
  const float scale = 0.17677669529663689f;  // 1/sqrt(32)
  v8f s0 = wmma16(a_q, bk0, vzero8());
  v8f s1 = wmma16(a_q, bk1, vzero8());

  float alpha[8];
#pragma unroll
  for (int i = 0; i < 8; ++i) {
    s0[i] *= scale;
    s1[i] *= scale;
    float m = fmaxf(s0[i], s1[i]);
    m = fmaxf(m, __shfl_xor(m, 1, 32));
    m = fmaxf(m, __shfl_xor(m, 2, 32));
    m = fmaxf(m, __shfl_xor(m, 4, 32));
    m = fmaxf(m, __shfl_xor(m, 8, 32));
    float nm = fmaxf(rmax[i], m);
    alpha[i] = __expf(rmax[i] - nm);
    rmax[i] = nm;
  }
#pragma unroll
  for (int i = 0; i < 8; ++i) {
    float p0 = __expf(s0[i] - rmax[i]);
    float p1 = __expf(s1[i] - rmax[i]);
    float sm = p0 + p1;
    sm += __shfl_xor(sm, 1, 32);
    sm += __shfl_xor(sm, 2, 32);
    sm += __shfl_xor(sm, 4, 32);
    sm += __shfl_xor(sm, 8, 32);
    rsum[i] = rsum[i] * alpha[i] + sm;
    acc0[i] *= alpha[i];
    acc1[i] *= alpha[i];
    stg[i + 8 * half][l16]      = (_Float16)p0;  // C-layout -> [m][key]
    stg[i + 8 * half][16 + l16] = (_Float16)p1;
  }
  asm volatile("s_wait_dscnt 0" ::: "memory");  // wave-private staging, no barrier

  // reload P as A-fragment (transpose via LDS), then accumulate P @ zv
  v16h a_p = load_frag(&stg[l16][half * 8]);
  acc0 = wmma16(a_p, bv0, acc0);
  acc1 = wmma16(a_p, bv1, acc1);
}

__global__ __launch_bounds__(256) void attn_kernel(
    const _Float16* __restrict__ zq,   // [B,H,T,rank]
    const _Float16* __restrict__ zk,   // [B,H,T,rank]
    const _Float16* __restrict__ zvT,  // [B,H,rank,T]
    _Float16* __restrict__ yr) {       // [B,T,H*rank]
  __shared__ __align__(16) _Float16 stage[8][16][40];  // per-wave P transpose tile

  const int lane = threadIdx.x & 31;
  const int wid  = threadIdx.x >> 5;
  const int gw   = blockIdx.x * 8 + wid;
  const int qt   = gw & 127;   // 128 query tiles per (b,h)
  const int bh   = gw >> 7;    // 0..31
  const int half = lane >> 4, l16 = lane & 15;

  const _Float16* zq_bh = zq + (size_t)bh * T_SEQ * RANK;
  const _Float16* zk_bh = zk + (size_t)bh * T_SEQ * RANK;
  const _Float16* zv_bh = zvT + (size_t)bh * RANK * T_SEQ;
  _Float16 (*stg)[40] = stage[wid];

  const int q0 = qt * 16;
  const v16h a_q = load_frag(zq_bh + (size_t)(q0 + l16) * RANK + half * 8);

  v8f acc0 = vzero8(), acc1 = vzero8();
  float rmax[8], rsum[8];
#pragma unroll
  for (int i = 0; i < 8; ++i) { rmax[i] = -3.0e38f; rsum[i] = 0.0f; }

  v16h k0P, k1P, v0P, v1P, k0Q, k1Q, v0Q, v1Q;
  load_kv(zk_bh, zv_bh, 0, half, l16, k0P, k1P, v0P, v1P);
  for (int kb = 0; kb < T_SEQ; kb += 64) {
    load_kv(zk_bh, zv_bh, kb + 32, half, l16, k0Q, k1Q, v0Q, v1Q);
    attn_step(a_q, k0P, k1P, v0P, v1P, acc0, acc1, rmax, rsum, stg, half, l16);
    int kn = (kb + 64) & (T_SEQ - 1);  // wrapped redundant prefetch on final iter
    load_kv(zk_bh, zv_bh, kn, half, l16, k0P, k1P, v0P, v1P);
    attn_step(a_q, k0Q, k1Q, v0Q, v1Q, acc0, acc1, rmax, rsum, stg, half, l16);
  }

  // epilogue: normalize and store yr[b][t][h*32 + r] (flat [4096 x 512])
  const int b = bh >> 4, h = bh & 15;
#pragma unroll
  for (int i = 0; i < 8; ++i) {
    int m = i + 8 * half;
    int t = q0 + m;
    float inv = 1.0f / rsum[i];
    size_t base = ((size_t)b * T_SEQ + t) * N_LR + h * RANK;
    yr[base + l16]      = (_Float16)(acc0[i] * inv);
    yr[base + 16 + l16] = (_Float16)(acc1[i] * inv);
  }
}

// ---------------------------------------------------------------------------
// output GEMM: out[4096x1024] = Yr[4096x512] @ Wuo_nmajor^T + bo  (f32 out)
// ---------------------------------------------------------------------------
__global__ __launch_bounds__(256) void out_gemm(
    const _Float16* __restrict__ A, const _Float16* __restrict__ Wn,
    const float* __restrict__ bo, float* __restrict__ out) {
  const int lane = threadIdx.x & 31;
  const int wid  = threadIdx.x >> 5;
  const int gw   = blockIdx.x * 8 + wid;
  const int wn   = gw & 15;   // 16 n-tiles of 64 (N=1024)
  const int wm   = gw >> 4;   // 256 m-tiles of 16
  const int half = lane >> 4, l16 = lane & 15;
  const int mbase = wm * 16, nbase = wn * 64;
  const int arow = mbase + l16;

  v8f acc[4];
#pragma unroll
  for (int t = 0; t < 4; ++t) acc[t] = vzero8();

  v16h aP, bP[4], aQ, bQ[4];
  load_ab(A, Wn, arow, nbase, 0, N_LR, half, l16, aP, bP);
  for (int k0 = 0; k0 < N_LR; k0 += 64) {
    load_ab(A, Wn, arow, nbase, k0 + 32, N_LR, half, l16, aQ, bQ);
#pragma unroll
    for (int t = 0; t < 4; ++t) acc[t] = wmma16(aP, bP[t], acc[t]);
    int k2 = k0 + 64;
    if (k2 >= N_LR) k2 = 0;  // wrapped redundant prefetch on final iter
    load_ab(A, Wn, arow, nbase, k2, N_LR, half, l16, aP, bP);
#pragma unroll
    for (int t = 0; t < 4; ++t) acc[t] = wmma16(aQ, bQ[t], acc[t]);
  }

#pragma unroll
  for (int t = 0; t < 4; ++t) {
    int n = nbase + t * 16 + l16;
    float bias = bo[n];
#pragma unroll
    for (int i = 0; i < 8; ++i) {
      int mi = mbase + i + 8 * half;
      out[(size_t)mi * D_MODEL + n] = acc[t][i] + bias;
    }
  }
}

// ---------------------------------------------------------------------------
// host launcher
// ---------------------------------------------------------------------------
extern "C" void kernel_launch(void* const* d_in, const int* in_sizes, int n_in,
                              void* d_out, int out_size, void* d_ws, size_t ws_size,
                              hipStream_t stream) {
  const float* x  = (const float*)d_in[0];
  const float* Wq = (const float*)d_in[1];
  const float* bq = (const float*)d_in[2];
  const float* Wk = (const float*)d_in[3];
  const float* bk = (const float*)d_in[4];
  const float* Wv = (const float*)d_in[5];
  const float* bv = (const float*)d_in[6];
  const float* Wo = (const float*)d_in[7];
  const float* bo = (const float*)d_in[8];
  const float* U  = (const float*)d_in[9];
  const float* V  = (const float*)d_in[10];

  char* ws = (char*)d_ws;
  size_t off = 0;
  auto take = [&](size_t bytes) {
    char* p = ws + off;
    off = (off + bytes + 255) & ~(size_t)255;
    return p;
  };
  _Float16* x16 = (_Float16*)take((size_t)M_ROWS * D_MODEL * 2);  // 8 MB
  _Float16* wqv = (_Float16*)take((size_t)N_LR * D_MODEL * 2);    // 1 MB
  _Float16* wkv = (_Float16*)take((size_t)N_LR * D_MODEL * 2);
  _Float16* wvv = (_Float16*)take((size_t)N_LR * D_MODEL * 2);
  _Float16* wuo = (_Float16*)take((size_t)D_MODEL * N_LR * 2);
  float*    zbq = (float*)take(N_LR * 4);
  float*    zbk = (float*)take(N_LR * 4);
  float*    zbv = (float*)take(N_LR * 4);
  _Float16* zq  = (_Float16*)take((size_t)M_ROWS * N_LR * 2);     // 4 MB [B,H,T,r]
  _Float16* zk  = (_Float16*)take((size_t)M_ROWS * N_LR * 2);
  _Float16* zvT = (_Float16*)take((size_t)M_ROWS * N_LR * 2);     //      [B,H,r,T]
  _Float16* yr  = (_Float16*)take((size_t)M_ROWS * N_LR * 2);     //      [B*T, H*r]

  // stage 1: pack + fold weights
  cvt_f32_f16<<<(M_ROWS * D_MODEL) / 256, 256, 0, stream>>>(x, x16,
                                                            M_ROWS * D_MODEL);
  make_wproj<<<(N_LR * D_MODEL) / 256, 256, 0, stream>>>(Wq, V, wqv);
  make_wproj<<<(N_LR * D_MODEL) / 256, 256, 0, stream>>>(Wk, V, wkv);
  make_wproj<<<(N_LR * D_MODEL) / 256, 256, 0, stream>>>(Wv, V, wvv);
  make_wout<<<(D_MODEL * N_LR) / 256, 256, 0, stream>>>(Wo, U, wuo);
  make_pbias<<<2, 256, 0, stream>>>(bq, V, zbq);
  make_pbias<<<2, 256, 0, stream>>>(bk, V, zbk);
  make_pbias<<<2, 256, 0, stream>>>(bv, V, zbv);

  // stage 2: fused QKV + low-rank projection (WMMA GEMMs)
  proj_gemm<<<256, 256, 0, stream>>>(x16, wqv, zbq, zq, 0);
  proj_gemm<<<256, 256, 0, stream>>>(x16, wkv, zbk, zk, 0);
  proj_gemm<<<256, 256, 0, stream>>>(x16, wvv, zbv, zvT, 1);

  // stage 3: flash attention in rank space (software-pipelined kv loads)
  attn_kernel<<<512, 256, 0, stream>>>(zq, zk, zvT, yr);

  // stage 4: fused U-reconstruct + output projection
  out_gemm<<<512, 256, 0, stream>>>(yr, wuo, bo, (float*)d_out);
}
